// GeneralPointSetAbstraction_31576599560743
// MI455X (gfx1250) — matrix operand order repaired and checked by reading, not compiled, not run on hardware
//
#include <hip/hip_runtime.h>
#include <hip/hip_bf16.h>

typedef __attribute__((ext_vector_type(16))) _Float16 v16h;
typedef __attribute__((ext_vector_type(8)))  _Float16 v8h;
typedef __attribute__((ext_vector_type(8)))  float    v8f;

#define BATCH 4
#define NPTS  16384
#define MCENT 2048
#define CFEAT 64
#define NS    32
#define H1DIM 64
#define H2DIM 128
#define KPAD  96
#define R2    0.64f

#define OUT_FEATS (BATCH*MCENT*3)
#define OUT_BID   (OUT_FEATS + BATCH*MCENT*H2DIM)

__device__ __forceinline__ unsigned long long wave_max_u64(unsigned long long v) {
#pragma unroll
  for (int o = 16; o > 0; o >>= 1) {
    unsigned lo = __shfl_xor((unsigned)v, o, 32);
    unsigned hi = __shfl_xor((unsigned)(v >> 32), o, 32);
    unsigned long long w = ((unsigned long long)hi << 32) | lo;
    if (w > v) v = w;
  }
  return v;
}

// ---------------- Kernel 1: farthest point sampling ----------------
// One 1024-thread block per batch. Each thread owns 16 points in registers.
// Argmax packed as (dist_bits << 32) | (N-1-idx) so u64-max == first-occurrence argmax.
__global__ __launch_bounds__(1024) void fps_kernel(const float* __restrict__ xyz,
                                                   float* out) {
  __shared__ unsigned long long wred[32];
  const int b    = blockIdx.x;
  const int tid  = threadIdx.x;
  const int lane = tid & 31, wid = tid >> 5;
  const float* bx = xyz + (size_t)b * NPTS * 3;

  float px[16], py[16], pz[16], dist[16];
#pragma unroll
  for (int i = 0; i < 16; ++i) {
    int p = tid + i * 1024;
    px[i] = bx[p*3+0]; py[i] = bx[p*3+1]; pz[i] = bx[p*3+2];
    dist[i] = 1.0e10f;
  }

  int last = 0;
  if (tid == 0) {
    out[(b*MCENT+0)*3+0] = bx[0];
    out[(b*MCENT+0)*3+1] = bx[1];
    out[(b*MCENT+0)*3+2] = bx[2];
  }

  for (int it = 1; it < MCENT; ++it) {
    int ls = __builtin_amdgcn_readfirstlane(last);
    float cx = bx[ls*3+0], cy = bx[ls*3+1], cz = bx[ls*3+2];
    float bd = -1.0f; int bp = 0;
#pragma unroll
    for (int i = 0; i < 16; ++i) {
      float dx = px[i]-cx, dy = py[i]-cy, dz = pz[i]-cz;
      float d  = dx*dx + dy*dy + dz*dz;
      float nd = dist[i] < d ? dist[i] : d;
      dist[i] = nd;
      if (nd > bd) { bd = nd; bp = tid + i*1024; }
    }
    unsigned long long pk =
        ((unsigned long long)__float_as_uint(bd) << 32) | (unsigned)(NPTS - 1 - bp);
    pk = wave_max_u64(pk);
    if (lane == 0) wred[wid] = pk;
    __syncthreads();
    pk = wave_max_u64(wred[lane]);
    __syncthreads();
    last = NPTS - 1 - (int)(unsigned)pk;
    if (tid == 0) {
      out[(b*MCENT+it)*3+0] = bx[last*3+0];
      out[(b*MCENT+it)*3+1] = bx[last*3+1];
      out[(b*MCENT+it)*3+2] = bx[last*3+2];
    }
  }
}

// ------- Kernel 2: ball query + gather + 2-layer MLP (WMMA) + maxpool -------
// 128 threads = 4 waves per block; one wave per centroid.
__global__ __launch_bounds__(128) void sa_kernel(const float* __restrict__ xyz,
                                                 const float* __restrict__ feats,
                                                 const float* __restrict__ w1,
                                                 const float* __restrict__ b1,
                                                 const float* __restrict__ w2,
                                                 const float* __restrict__ b2,
                                                 float* out) {
  __shared__ __attribute__((aligned(32))) _Float16 W1c[H1DIM][KPAD];   // [n][k], transposed
  __shared__ __attribute__((aligned(32))) _Float16 W2c[H2DIM][H1DIM];  // [n][k], transposed
  __shared__ float b1s[H1DIM];
  __shared__ float b2s[H2DIM];
  __shared__ __attribute__((aligned(32))) _Float16 Xs[4][NS][KPAD];
  __shared__ __attribute__((aligned(32))) _Float16 H1s[4][NS][H1DIM];
  __shared__ int sel[4][NS];

  const int tid = threadIdx.x;
  // Stage weights (transposed) + biases once per block.
  for (int i = tid; i < H1DIM*KPAD; i += 128) {
    int n = i / KPAD, k = i % KPAD;
    W1c[n][k] = (k < 67) ? (_Float16)w1[k*H1DIM + n] : (_Float16)0.0f;
  }
  for (int i = tid; i < H2DIM*H1DIM; i += 128) {
    int n = i / H1DIM, k = i % H1DIM;
    W2c[n][k] = (_Float16)w2[k*H2DIM + n];
  }
  if (tid < H1DIM) b1s[tid] = b1[tid];
  b2s[tid] = b2[tid];
  __syncthreads();

  const int wid  = tid >> 5, lane = tid & 31;
  const int cent = blockIdx.x * 4 + wid;            // [0, B*MCENT)
  const int b    = cent / MCENT;
  const float* bx = xyz + (size_t)b * NPTS * 3;
  const float cx = out[cent*3+0], cy = out[cent*3+1], cz = out[cent*3+2];

  // ---- ball query: first NS points (index order) with d2 < R2 ----
  int cnt = 0;
  for (int base = 0; base < NPTS && cnt < NS; base += 32) {
    int p = base + lane;
    float dx = bx[p*3+0]-cx, dy = bx[p*3+1]-cy, dz = bx[p*3+2]-cz;
    float d2 = dx*dx + dy*dy + dz*dz;
    bool hit = d2 < R2;
    unsigned msk = (unsigned)__ballot(hit);
    int pos = cnt + __popc(msk & ((1u << lane) - 1u));
    if (hit && pos < NS) sel[wid][pos] = p;
    cnt += __popc(msk);
  }
  asm volatile("s_wait_dscnt 0" ::: "memory");
  if (cnt > NS) cnt = NS;
  const int si = sel[wid][lane < cnt ? lane : 0];   // fill empty slots with first

  // ---- gather: build X row (rel xyz ++ feats, padded to KPAD) in f16 ----
  Xs[wid][lane][0] = (_Float16)(bx[si*3+0]-cx);
  Xs[wid][lane][1] = (_Float16)(bx[si*3+1]-cy);
  Xs[wid][lane][2] = (_Float16)(bx[si*3+2]-cz);
  {
    const float* fs = feats + ((size_t)b*NPTS + si) * CFEAT;
#pragma unroll 8
    for (int j = 0; j < CFEAT; ++j) Xs[wid][lane][3+j] = (_Float16)fs[j];
#pragma unroll
    for (int j = 67; j < KPAD; ++j) Xs[wid][lane][j] = (_Float16)0.0f;
  }
  asm volatile("s_wait_dscnt 0" ::: "memory");

  const int halfrow = lane >> 4;   // 0: lanes 0-15, 1: lanes 16-31
  const int lcol    = lane & 15;

  // ---- layer 1: H1 = relu(X @ W1 + b1), via 16x16x32 f16 WMMA ----
#pragma unroll
  for (int nt = 0; nt < 4; ++nt) {
#pragma unroll
    for (int mt = 0; mt < 2; ++mt) {
      v8f acc = {};
#pragma unroll
      for (int kt = 0; kt < 3; ++kt) {
        const _Float16* ap = &Xs[wid][mt*16 + lcol][kt*32 + halfrow*16];
        v16h a = *(const v16h*)ap;
        v16h bb = *(const v16h*)&W1c[nt*16 + lcol][kt*32 + halfrow*16];
        acc = __builtin_amdgcn_wmma_f32_16x16x32_f16(false, a, false, bb,
                                                     (short)0, acc, false, false);
      }
      float bias = b1s[nt*16 + lcol];
#pragma unroll
      for (int v = 0; v < 8; ++v) {
        float h = acc[v] + bias;
        h = h > 0.0f ? h : 0.0f;
        H1s[wid][mt*16 + halfrow*8 + v][nt*16 + lcol] = (_Float16)h;
      }
    }
  }
  asm volatile("s_wait_dscnt 0" ::: "memory");

  // ---- layer 2: relu(H1 @ W2 + b2), fused max over the 32 samples ----
  float* ofeat = out + OUT_FEATS + (size_t)cent * H2DIM;
#pragma unroll
  for (int nt = 0; nt < 8; ++nt) {
    float mx = 0.0f;                  // relu output >= 0
#pragma unroll
    for (int mt = 0; mt < 2; ++mt) {
      v8f acc = {};
#pragma unroll
      for (int kt = 0; kt < 2; ++kt) {
        const _Float16* ap = &H1s[wid][mt*16 + lcol][kt*32 + halfrow*16];
        v16h a = *(const v16h*)ap;
        v16h bb = *(const v16h*)&W2c[nt*16 + lcol][kt*32 + halfrow*16];
        acc = __builtin_amdgcn_wmma_f32_16x16x32_f16(false, a, false, bb,
                                                     (short)0, acc, false, false);
      }
      float bias = b2s[nt*16 + lcol];
#pragma unroll
      for (int v = 0; v < 8; ++v) {
        float h = acc[v] + bias;
        h = h > 0.0f ? h : 0.0f;
        mx = fmaxf(mx, h);
      }
    }
    mx = fmaxf(mx, __shfl_xor(mx, 16, 32));   // rows 0-7 vs 8-15 halves
    if (lane < 16) ofeat[nt*16 + lcol] = mx;
  }
  if (lane == 0) out[OUT_BID + cent] = 0.0f;  // int32 zero == 0.0f bit pattern
}

extern "C" void kernel_launch(void* const* d_in, const int* in_sizes, int n_in,
                              void* d_out, int out_size, void* d_ws, size_t ws_size,
                              hipStream_t stream) {
  const float* xyz   = (const float*)d_in[0];
  const float* feats = (const float*)d_in[1];
  // d_in[2] = bid (all zeros, unused)
  const float* w1 = (const float*)d_in[3];
  const float* b1 = (const float*)d_in[4];
  const float* w2 = (const float*)d_in[5];
  const float* b2 = (const float*)d_in[6];
  float* out = (float*)d_out;

  fps_kernel<<<BATCH, 1024, 0, stream>>>(xyz, out);
  sa_kernel<<<(BATCH*MCENT)/4, 128, 0, stream>>>(xyz, feats, w1, b1, w2, b2, out);
}
